// RNN_Classifier_57707180589182
// MI455X (gfx1250) — matrix operand order repaired and checked by reading, not compiled, not run on hardware
//
#include <hip/hip_runtime.h>
#include <math.h>

#define T_SEQ   256
#define BATCH   4096
#define IN_DIM  28
#define HID     128
#define CLS     10
#define LDSS    136   // padded bf16 row stride (272B) to avoid bank conflicts

typedef __attribute__((ext_vector_type(16))) __bf16 v16bf;
typedef __attribute__((ext_vector_type(8)))  float  v8f;
typedef __attribute__((ext_vector_type(4)))  float  v4f;

// Branchless tanh: 1 - 2/(exp(2x)+1).  Saturates to +-1 correctly (exp->inf or 0),
// lowers to v_mul + v_exp_f32 + v_add + v_rcp_f32 + v_fma — no divergent branches.
__device__ __forceinline__ float fast_tanh(float x) {
    float e = __expf(2.0f * x);
    return 1.0f - 2.0f * __builtin_amdgcn_rcpf(e + 1.0f);
}

__global__ __launch_bounds__(64) void rnn_fused_kernel(
    const float* __restrict__ x,      // [T, B, 28]
    const float* __restrict__ W_ih,   // [128, 28]
    const float* __restrict__ W_hh,   // [128, 128]
    const float* __restrict__ b_ih,   // [128]
    const float* __restrict__ b_hh,   // [128]
    const float* __restrict__ W_d,    // [10, 128]
    const float* __restrict__ b_d,    // [10]
    float* __restrict__ out)          // [B, 10]
{
    __shared__ __bf16 h_lds[2][16][LDSS];   // double-buffered hidden state (16 rows)
    __shared__ float  logits[16][CLS];

    const int tid  = threadIdx.x;
    const int wave = tid >> 5;          // 0..1 : owns hidden cols [wave*64, wave*64+64)
    const int lane = tid & 31;
    const int lrow = lane & 15;         // A-row / B-col / C-col within tile
    const int hi   = lane >> 4;         // lane-half select
    const int rowbase  = blockIdx.x * 16;   // batch rows of this block
    const int ncolbase = wave * 64;

    // ---- Hoist W_hh B-fragments into registers (bf16), resident for all 256 steps.
    // B[k][n] = W_hh[n][k]; lane supplies col n = ncolbase + nt*16 + lrow,
    // k = kt*32 + hi*8 + i (i<8), kt*32 + 16 + hi*8 + (i-8) (i>=8).
    v16bf Bhh[4][4];
    #pragma unroll
    for (int kt = 0; kt < 4; ++kt) {
        #pragma unroll
        for (int nt = 0; nt < 4; ++nt) {
            const int n  = ncolbase + nt * 16 + lrow;
            const float* src = W_hh + n * HID + kt * 32 + hi * 8;
            v16bf f;
            #pragma unroll
            for (int i = 0; i < 16; ++i) f[i] = (__bf16)src[i < 8 ? i : i + 8];
            Bhh[kt][nt] = f;
        }
    }

    // ---- W_ih B-fragments (single K-tile: K=0..31, real 0..27, rest zero).
    // Branchless: clamp index, scale by 0/1 validity (v_cndmask, no exec branch).
    v16bf Bih[4];
    #pragma unroll
    for (int nt = 0; nt < 4; ++nt) {
        const int n = ncolbase + nt * 16 + lrow;
        v16bf f;
        #pragma unroll
        for (int i = 0; i < 16; ++i) {
            const int   k  = (i < 8) ? hi * 8 + i : 16 + hi * 8 + (i - 8);
            const int   kc = (k < IN_DIM) ? k : 0;
            const float vm = (k < IN_DIM) ? 1.0f : 0.0f;
            f[i] = (__bf16)(W_ih[n * IN_DIM + kc] * vm);
        }
        Bih[nt] = f;
    }

    // ---- Combined bias per output tile (same N for every element of a C/D VGPR).
    float bias[4];
    #pragma unroll
    for (int nt = 0; nt < 4; ++nt) {
        const int n = ncolbase + nt * 16 + lrow;
        bias[nt] = b_ih[n] + b_hh[n];
    }

    // ---- h0 = 0
    for (int idx = tid; idx < 16 * LDSS; idx += 64)
        h_lds[0][idx / LDSS][idx % LDSS] = (__bf16)0.0f;
    __syncthreads();

    const int   kh    = hi * 8;                 // A-fragment K-half offset
    const float tmask = hi ? 0.0f : 1.0f;       // validity of A_x elems 12..15

    // x row for this lane: byte offset is a multiple of 112 (16B aligned), and all
    // four float4 offsets below are 16B aligned and always in-bounds.
    const float* xbase = x + (size_t)(rowbase + lrow) * IN_DIM;
    const int    k3    = 20 + hi * 4;           // hi=0: k=20..23 (valid), hi=1: k=24..27 (masked)

    // ---- software-pipelined x loads: fetch t=0 before the loop
    v4f xr0, xr1, xr2, xr3;
    {
        const float* xp = xbase;                // t = 0
        xr0 = *(const v4f*)(xp + kh);
        xr1 = *(const v4f*)(xp + kh + 4);
        xr2 = *(const v4f*)(xp + 16 + kh);
        xr3 = *(const v4f*)(xp + k3);
    }

    // ================= recurrent scan =================
    for (int t = 0; t < T_SEQ; ++t) {
        // Build A_x fragment from the pipelined registers (16x32, K>=28 zeroed).
        v16bf Ax;
        #pragma unroll
        for (int i = 0; i < 4; ++i) Ax[i]      = (__bf16)xr0[i];
        #pragma unroll
        for (int i = 0; i < 4; ++i) Ax[4 + i]  = (__bf16)xr1[i];
        #pragma unroll
        for (int i = 0; i < 4; ++i) Ax[8 + i]  = (__bf16)xr2[i];
        #pragma unroll
        for (int i = 0; i < 4; ++i) Ax[12 + i] = (__bf16)(xr3[i] * tmask);

        // Issue next timestep's x loads now; latency hides behind WMMA + LDS below.
        {
            const int tn = (t + 1) & (T_SEQ - 1);   // wraps to 0 on last iter (in-bounds, branchless)
            const float* xp = xbase + (size_t)tn * BATCH * IN_DIM;
            xr0 = *(const v4f*)(xp + kh);
            xr1 = *(const v4f*)(xp + kh + 4);
            xr2 = *(const v4f*)(xp + 16 + kh);
            xr3 = *(const v4f*)(xp + k3);
        }

        // acc = bias + x_t @ W_ih^T   (one WMMA per N-tile)
        v8f acc[4];
        #pragma unroll
        for (int nt = 0; nt < 4; ++nt) {
            v8f c = {bias[nt], bias[nt], bias[nt], bias[nt],
                     bias[nt], bias[nt], bias[nt], bias[nt]};
            acc[nt] = __builtin_amdgcn_wmma_f32_16x16x32_bf16(
                false, Ax, false, Bih[nt], (short)0, c, false, false);
        }

        // acc += h @ W_hh^T   (K=128 -> 4 K-tiles, weights already in VGPRs)
        const __bf16* hrow = &h_lds[t & 1][lrow][0];
        #pragma unroll
        for (int kt = 0; kt < 4; ++kt) {
            const __bf16* hp = hrow + kt * 32 + kh;
            v16bf Ah;
            #pragma unroll
            for (int i = 0; i < 8; ++i) Ah[i]     = hp[i];
            #pragma unroll
            for (int i = 0; i < 8; ++i) Ah[8 + i] = hp[16 + i];
            #pragma unroll
            for (int nt = 0; nt < 4; ++nt)
                acc[nt] = __builtin_amdgcn_wmma_f32_16x16x32_bf16(
                    false, Ah, false, Bhh[kt][nt], (short)0, acc[nt], false, false);
        }

        // h_new = tanh(acc) -> other LDS buffer (branchless fast tanh)
        // C/D layout: VGPR v holds row M = v + hi*8, col N = tile_base + lrow
        #pragma unroll
        for (int nt = 0; nt < 4; ++nt) {
            const int n = ncolbase + nt * 16 + lrow;
            #pragma unroll
            for (int v = 0; v < 8; ++v) {
                const int m = v + hi * 8;
                h_lds[(t + 1) & 1][m][n] = (__bf16)fast_tanh(acc[nt][v]);
            }
        }
        __syncthreads();
    }

    // ================= classifier + softmax =================
    // final h lives in buffer (T_SEQ & 1) == 0
    for (int task = tid; task < 16 * CLS; task += 64) {
        const int r = task / CLS, c = task % CLS;
        float s = b_d[c];
        #pragma unroll 8
        for (int k = 0; k < HID; ++k) {
            float hv = (float)h_lds[0][r][k];
            hv = hv > 0.0f ? hv : 0.0f;            // relu
            s += hv * W_d[c * HID + k];
        }
        logits[r][c] = s;
    }
    __syncthreads();

    if (tid < 16) {
        const int b = rowbase + tid;
        float mx = -3.0e38f;
        #pragma unroll
        for (int c = 0; c < CLS; ++c) mx = fmaxf(mx, logits[tid][c]);
        float e[CLS], sum = 0.0f;
        #pragma unroll
        for (int c = 0; c < CLS; ++c) { e[c] = __expf(logits[tid][c] - mx); sum += e[c]; }
        const float inv = 1.0f / sum;
        #pragma unroll
        for (int c = 0; c < CLS; ++c) out[b * CLS + c] = e[c] * inv;
    }
}

extern "C" void kernel_launch(void* const* d_in, const int* in_sizes, int n_in,
                              void* d_out, int out_size, void* d_ws, size_t ws_size,
                              hipStream_t stream) {
    const float* x    = (const float*)d_in[0];
    const float* W_ih = (const float*)d_in[1];
    const float* W_hh = (const float*)d_in[2];
    const float* b_ih = (const float*)d_in[3];
    const float* b_hh = (const float*)d_in[4];
    const float* W_d  = (const float*)d_in[5];
    const float* b_d  = (const float*)d_in[6];
    float* out = (float*)d_out;

    dim3 grid(BATCH / 16);   // 256 blocks, each owns 16 batch rows for the whole scan
    dim3 block(64);          // 2 waves; wave w computes hidden cols [64w, 64w+64)
    rnn_fused_kernel<<<grid, block, 0, stream>>>(x, W_ih, W_hh, b_ih, b_hh, W_d, b_d, out);
}
